// LinearAttention_58952721105326
// MI455X (gfx1250) — matrix operand order repaired and checked by reading, not compiled, not run on hardware
//
#include <hip/hip_runtime.h>

// ---------------------------------------------------------------------------
// LinearAttention for MI455X (gfx1250): all GEMMs via v_wmma_f32_16x16x32_bf16,
// async global->LDS staging in the projection kernel.
// B=8, C=CALC=512, H=8, HD=64, TQ=TK=2048
// ---------------------------------------------------------------------------

typedef __attribute__((ext_vector_type(16))) __bf16 v16bf;
typedef __attribute__((ext_vector_type(8)))  __bf16 v8bf;
typedef __attribute__((ext_vector_type(8)))  float  v8f;

#define BB 8
#define CD 512
#define TD 2048
#define NH 8

__device__ __forceinline__ v8f wmma_bf16(v16bf a, v16bf b, v8f c) {
  // (neg_a, A, neg_b, B, c_mod, C, reuse_a, reuse_b)
  return __builtin_amdgcn_wmma_f32_16x16x32_bf16(false, a, false, b,
                                                 (short)0, c, false, false);
}

// A-fragment (16x32 bf16, contiguous k = 16h + s): per lane two contiguous 8-element
// chunks at +0 and +8 (base already includes koffA = lane<16 ? 0 : 16).
__device__ __forceinline__ v16bf load_a_bf16(const __bf16* __restrict__ p) {
  v8bf lo = *(const v8bf*)(p);
  v8bf hi = *(const v8bf*)(p + 8);
  v16bf a;
#pragma unroll
  for (int j = 0; j < 8; ++j) { a[j] = lo[j]; a[8 + j] = hi[j]; }
  return a;
}

// B-fragment: 16 contiguous k-values (f32 source), converted to bf16
__device__ __forceinline__ v16bf load_b_f32(const float* __restrict__ p) {
  v16bf r;
#pragma unroll
  for (int j = 0; j < 16; ++j) r[j] = (__bf16)p[j];
  return r;
}

// ---------------------------------------------------------------------------
// prep_w: one-shot f32 -> bf16 conversion of the four weight matrices.
// grid (CD*CD/256, 4), block 256.
// ---------------------------------------------------------------------------
__global__ __launch_bounds__(256) void prep_w(
    const float* __restrict__ w0, const float* __restrict__ w1,
    const float* __restrict__ w2, const float* __restrict__ w3,
    __bf16* __restrict__ dst) {
  const float* s;
  switch (blockIdx.y) {
    case 0:  s = w0; break;
    case 1:  s = w1; break;
    case 2:  s = w2; break;
    default: s = w3; break;
  }
  int i = blockIdx.x * 256 + threadIdx.x;
  dst[(size_t)blockIdx.y * CD * CD + i] = (__bf16)s[i];
}

// ---------------------------------------------------------------------------
// branch_kernel: Y = act(LN_ch(W @ X + bias)) for one 512 x 32 column tile.
//   mode 0 (q): per-head softmax over HD, * 8^-0.5, write t-major (qT[b][t][c])
//   mode 1 (k): mask -> 0 else exp(.), write c-major (kexp[b][c][t])
//   mode 2 (v): * 1/64, write c-major (vfeat[b][c][t])
// grid (TD/32, B), block 256 (8 waves); wave w owns rows [64w, 64w+64).
// ---------------------------------------------------------------------------
__global__ __launch_bounds__(256) void branch_kernel(
    const float* __restrict__ X, const __bf16* __restrict__ W,
    const float* __restrict__ bias, const float* __restrict__ gamma,
    const float* __restrict__ beta, const unsigned char* __restrict__ mask,
    __bf16* __restrict__ outp, int mode) {
  __shared__ float Smem[32 * CD];     // 64KB: Xs(bf16) then Ys(f32), barrier-separated
  __shared__ float red[32][8];
  __shared__ float red2[32][8];
  __shared__ float stats[32][2];
  __bf16* Xs = (__bf16*)Smem;
  float*  Ys = Smem;

  const int tid  = threadIdx.x;
  const int b    = blockIdx.y;
  const int t0   = blockIdx.x * 32;
  const int lane = tid & 31;
  const int wave = tid >> 5;

  // stage X[c, t0:t0+32] -> Xs[t][c] (bf16); float4 global reads along t
  for (int i = tid; i < 32 * CD / 4; i += 256) {
    int c = i >> 3, t4 = (i & 7) * 4;
    float4 x = *(const float4*)(X + ((long)b * CD + c) * TD + t0 + t4);
    Xs[(t4 + 0) * CD + c] = (__bf16)x.x;
    Xs[(t4 + 1) * CD + c] = (__bf16)x.y;
    Xs[(t4 + 2) * CD + c] = (__bf16)x.z;
    Xs[(t4 + 3) * CD + c] = (__bf16)x.w;
  }
  __syncthreads();

  const int m0     = wave * 64;
  const int col    = lane & 15;
  const int koffA  = (lane < 16) ? 0 : 16;
  const int koffB  = (lane < 16) ? 0 : 16;
  v8f acc[4][2] = {};

  for (int kk = 0; kk < CD; kk += 32) {
    v16bf afrag[4];
#pragma unroll
    for (int mt = 0; mt < 4; ++mt) {
      int row = m0 + mt * 16 + col;
      afrag[mt] = load_a_bf16(W + (long)row * CD + kk + koffA);
    }
    v16bf bfrag[2];
#pragma unroll
    for (int nt = 0; nt < 2; ++nt)
      bfrag[nt] = *(const v16bf*)(Xs + (nt * 16 + col) * CD + kk + koffB);
#pragma unroll
    for (int mt = 0; mt < 4; ++mt)
#pragma unroll
      for (int nt = 0; nt < 2; ++nt)
        acc[mt][nt] = wmma_bf16(afrag[mt], bfrag[nt], acc[mt][nt]);
  }
  __syncthreads();  // done with Xs; reuse Smem as Ys

  const int rowoff = (lane < 16) ? 0 : 8;
#pragma unroll
  for (int mt = 0; mt < 4; ++mt)
#pragma unroll
    for (int nt = 0; nt < 2; ++nt) {
      int t = nt * 16 + col;
#pragma unroll
      for (int r = 0; r < 8; ++r) {
        int row = m0 + mt * 16 + r + rowoff;
        Ys[t * CD + row] = acc[mt][nt][r] + bias[row];
      }
    }
  __syncthreads();

  // LayerNorm over channels (512) per t column: 8 workers / column
  {
    int t = tid >> 3, w = tid & 7;
    float s = 0.f, s2 = 0.f;
    for (int j = 0; j < 64; ++j) {
      float v = Ys[t * CD + w * 64 + j];
      s += v; s2 += v * v;
    }
    red[t][w] = s; red2[t][w] = s2;
  }
  __syncthreads();
  if (tid < 32) {
    float s = 0.f, s2 = 0.f;
    for (int w = 0; w < 8; ++w) { s += red[tid][w]; s2 += red2[tid][w]; }
    float mu  = s * (1.f / CD);
    float var = s2 * (1.f / CD) - mu * mu;
    stats[tid][0] = mu;
    stats[tid][1] = rsqrtf(var + 1e-5f);
  }
  __syncthreads();

  for (int i = tid; i < 32 * CD; i += 256) {
    int t = i >> 9, c = i & (CD - 1);
    float mu = stats[t][0], rstd = stats[t][1];
    float val = (Ys[i] - mu) * rstd * gamma[c] + beta[c];
    float a = (val > 0.f) ? (val + 1.f) : __expf(val);  // elu + 1
    if (mode == 1) {
      bool m = mask[(long)b * TD + t0 + t] != 0;
      outp[((long)b * CD + c) * TD + t0 + t] = (__bf16)(m ? 0.f : __expf(a));
    } else if (mode == 2) {
      outp[((long)b * CD + c) * TD + t0 + t] = (__bf16)(a * (1.f / 64.f));
    } else {
      Ys[i] = __expf(a);  // q: stash exp for softmax over HD
    }
  }
  if (mode == 0) {
    __syncthreads();
    {
      int t = tid >> 3, grp = tid & 7;  // 32 t x 8 head-groups of 64
      float s = 0.f;
      for (int j = 0; j < 64; ++j) s += Ys[t * CD + grp * 64 + j];
      red[t][grp] = 0.35355339059f / s;  // includes H^-0.5
    }
    __syncthreads();
    for (int i = tid; i < 32 * CD; i += 256) {
      int t = i >> 9, c = i & (CD - 1);
      outp[((long)b * TD + t0 + t) * CD + c] = (__bf16)(Ys[i] * red[t][c >> 6]);
    }
  }
}

// ---------------------------------------------------------------------------
// ctx_kernel: per (b,h): ctx[kd][vd] = (sum_t expk[kd,t] * v[vd,t]) / rowsum[kd]
// grid (H, B), block 128 (4 waves); wave = kd-tile row, K loop over t (2048).
// ---------------------------------------------------------------------------
__global__ __launch_bounds__(128) void ctx_kernel(
    const __bf16* __restrict__ kexp, const __bf16* __restrict__ vfeat,
    float* __restrict__ ctx) {
  __shared__ float rsum[64];
  const int tid = threadIdx.x, lane = tid & 31, rt = tid >> 5;
  const int h = blockIdx.x, b = blockIdx.y;
  if (tid < 64) rsum[tid] = 0.f;
  __syncthreads();

  const int col   = lane & 15;
  const int koffA = (lane < 16) ? 0 : 16;
  const int koffB = (lane < 16) ? 0 : 16;
  const __bf16* Abase = kexp + ((long)b * CD + h * 64 + rt * 16 + col) * TD;
  v8f acc[4] = {};
  float rs = 0.f;

  for (int tb = 0; tb < TD; tb += 32) {
    v16bf a = load_a_bf16(Abase + tb + koffA);
#pragma unroll
    for (int j = 0; j < 16; ++j) rs += (float)a[j];
#pragma unroll
    for (int vt = 0; vt < 4; ++vt) {
      const __bf16* bp = vfeat + ((long)b * CD + h * 64 + vt * 16 + col) * TD + tb + koffB;
      acc[vt] = wmma_bf16(a, *(const v16bf*)bp, acc[vt]);
    }
  }
  atomicAdd(&rsum[rt * 16 + col], rs);
  __syncthreads();

  const int rowoff = (lane < 16) ? 0 : 8;
#pragma unroll
  for (int vt = 0; vt < 4; ++vt)
#pragma unroll
    for (int r = 0; r < 8; ++r) {
      int kd = rt * 16 + r + rowoff;
      int vd = vt * 16 + col;
      ctx[(((long)b * NH + h) * 64 + kd) * 64 + vd] = acc[vt][r] / (rsum[kd] + 1e-30f);
    }
}

// ---------------------------------------------------------------------------
// mmat_kernel: M_b[o, h*64+kd] = sum_vd proj_w[o, h*64+vd] * ctx[b,h,kd,vd]
// grid (4, H, B), block 256 (8 waves): wave -> one 16-row o-tile x 4 kd-tiles.
// ---------------------------------------------------------------------------
__global__ __launch_bounds__(256) void mmat_kernel(
    const __bf16* __restrict__ projw, const float* __restrict__ ctx,
    __bf16* __restrict__ Mm) {
  const int tid = threadIdx.x, lane = tid & 31, wave = tid >> 5;
  const int h = blockIdx.y, b = blockIdx.z;
  const int o0 = blockIdx.x * 128 + wave * 16;
  const int col   = lane & 15;
  const int koffA = (lane < 16) ? 0 : 16;
  const int koffB = (lane < 16) ? 0 : 16;
  v8f acc[4] = {};

  for (int kk = 0; kk < 64; kk += 32) {
    v16bf a = load_a_bf16(projw + (long)(o0 + col) * CD + h * 64 + kk + koffA);
#pragma unroll
    for (int kt = 0; kt < 4; ++kt) {
      const float* bp = ctx + (((long)b * NH + h) * 64 + kt * 16 + col) * 64 + kk + koffB;
      acc[kt] = wmma_bf16(a, load_b_f32(bp), acc[kt]);
    }
  }
  const int rowoff = (lane < 16) ? 0 : 8;
#pragma unroll
  for (int kt = 0; kt < 4; ++kt)
#pragma unroll
    for (int r = 0; r < 8; ++r) {
      int o  = o0 + r + rowoff;
      int cc = h * 64 + kt * 16 + col;
      Mm[((long)b * CD + o) * CD + cc] = (__bf16)acc[kt][r];
    }
}

// ---------------------------------------------------------------------------
// proj_kernel: Y = LN_ch(M_b @ qT + proj_b + residual) -> d_out [B][C][TQ]
// grid (TD/32, B), block 256. qT tile staged with GLOBAL_LOAD_ASYNC_TO_LDS_B128
// (ASYNCcnt path), then fenced with s_wait_asynccnt + barrier.
// ---------------------------------------------------------------------------
__global__ __launch_bounds__(256) void proj_kernel(
    const __bf16* __restrict__ qT, const __bf16* __restrict__ Mm,
    const float* __restrict__ pb, const float* __restrict__ resid,
    const float* __restrict__ gamma, const float* __restrict__ beta,
    float* __restrict__ outp) {
  __shared__ float Smem[32 * CD];
  __shared__ float red[32][8];
  __shared__ float red2[32][8];
  __shared__ float stats[32][2];
  __bf16* Bs = (__bf16*)Smem;
  float*  Ys = Smem;

  const int tid  = threadIdx.x;
  const int b    = blockIdx.y;
  const int t0   = blockIdx.x * 32;
  const int lane = tid & 31;
  const int wave = tid >> 5;

  // async stage qT[b][t0:t0+32][0:512] (32KB contiguous bf16) -> LDS
  {
    unsigned long long saddr = (unsigned long long)(qT + ((long)b * TD + t0) * CD);
    unsigned lds_base = (unsigned)(unsigned long long)Bs;
#pragma unroll
    for (int j = 0; j < 8; ++j) {
      unsigned voff = (tid + j * 256) * 16;      // 2048 x 16B transfers
      unsigned lds  = lds_base + voff;
      asm volatile("global_load_async_to_lds_b128 %0, %1, %2"
                   :: "v"(lds), "v"(voff), "s"(saddr)
                   : "memory");
    }
    asm volatile("s_wait_asynccnt 0x0" ::: "memory");
  }
  __syncthreads();

  const int m0     = wave * 64;
  const int col    = lane & 15;
  const int koffA  = (lane < 16) ? 0 : 16;
  const int koffB  = (lane < 16) ? 0 : 16;
  v8f acc[4][2] = {};

  for (int kk = 0; kk < CD; kk += 32) {
    v16bf afrag[4];
#pragma unroll
    for (int mt = 0; mt < 4; ++mt) {
      int row = m0 + mt * 16 + col;
      afrag[mt] = load_a_bf16(Mm + ((long)b * CD + row) * CD + kk + koffA);
    }
    v16bf bfrag[2];
#pragma unroll
    for (int nt = 0; nt < 2; ++nt)
      bfrag[nt] = *(const v16bf*)(Bs + (nt * 16 + col) * CD + kk + koffB);
#pragma unroll
    for (int mt = 0; mt < 4; ++mt)
#pragma unroll
      for (int nt = 0; nt < 2; ++nt)
        acc[mt][nt] = wmma_bf16(afrag[mt], bfrag[nt], acc[mt][nt]);
  }
  __syncthreads();

  const int rowoff = (lane < 16) ? 0 : 8;
#pragma unroll
  for (int mt = 0; mt < 4; ++mt)
#pragma unroll
    for (int nt = 0; nt < 2; ++nt) {
      int t = nt * 16 + col;
#pragma unroll
      for (int r = 0; r < 8; ++r) {
        int row = m0 + mt * 16 + r + rowoff;
        Ys[t * CD + row] = acc[mt][nt][r] + pb[row] +
                           resid[((long)b * CD + row) * TD + t0 + t];
      }
    }
  __syncthreads();

  {
    int t = tid >> 3, w = tid & 7;
    float s = 0.f, s2 = 0.f;
    for (int j = 0; j < 64; ++j) {
      float v = Ys[t * CD + w * 64 + j];
      s += v; s2 += v * v;
    }
    red[t][w] = s; red2[t][w] = s2;
  }
  __syncthreads();
  if (tid < 32) {
    float s = 0.f, s2 = 0.f;
    for (int w = 0; w < 8; ++w) { s += red[tid][w]; s2 += red2[tid][w]; }
    float mu  = s * (1.f / CD);
    float var = s2 * (1.f / CD) - mu * mu;
    stats[tid][0] = mu;
    stats[tid][1] = rsqrtf(var + 1e-5f);
  }
  __syncthreads();

  for (int i = tid; i < 32 * CD; i += 256) {
    int t = i >> 9, c = i & (CD - 1);
    float y = (Ys[i] - stats[t][0]) * stats[t][1] * gamma[c] + beta[c];
    outp[((long)b * CD + c) * TD + t0 + t] = y;
  }
}

// ---------------------------------------------------------------------------
extern "C" void kernel_launch(void* const* d_in, const int* in_sizes, int n_in,
                              void* d_out, int out_size, void* d_ws, size_t ws_size,
                              hipStream_t stream) {
  (void)in_sizes; (void)n_in; (void)out_size; (void)ws_size;
  const float* queries = (const float*)d_in[0];
  const float* keys    = (const float*)d_in[1];
  const float* values  = (const float*)d_in[2];
  const unsigned char* mask = (const unsigned char*)d_in[3];
  const float* q_w = (const float*)d_in[4];
  const float* q_b = (const float*)d_in[5];
  const float* q_g = (const float*)d_in[6];
  const float* q_bb = (const float*)d_in[7];
  const float* k_w = (const float*)d_in[8];
  const float* k_b = (const float*)d_in[9];
  const float* k_g = (const float*)d_in[10];
  const float* k_bb = (const float*)d_in[11];
  const float* v_w = (const float*)d_in[12];
  const float* v_b = (const float*)d_in[13];
  const float* v_g = (const float*)d_in[14];
  const float* v_bb = (const float*)d_in[15];
  const float* proj_w = (const float*)d_in[16];
  const float* proj_b = (const float*)d_in[17];
  const float* norm_g = (const float*)d_in[18];
  const float* norm_b = (const float*)d_in[19];

  const size_t szFeat = (size_t)BB * CD * TD * sizeof(__bf16);  // 16 MB
  char* p = (char*)d_ws;
  __bf16* qT    = (__bf16*)p; p += szFeat;                      // [B][T][C]
  __bf16* kexp  = (__bf16*)p; p += szFeat;                      // [B][C][T]
  __bf16* vfeat = (__bf16*)p; p += szFeat;                      // [B][C][T]
  float*  ctxw  = (float*)p;  p += (size_t)BB * NH * 64 * 64 * sizeof(float);
  __bf16* Mm    = (__bf16*)p; p += (size_t)BB * CD * CD * sizeof(__bf16);
  __bf16* Wbf   = (__bf16*)p;                                   // 4 x [512][512]
  __bf16* wq = Wbf;
  __bf16* wk = Wbf + 1 * CD * CD;
  __bf16* wv = Wbf + 2 * CD * CD;
  __bf16* wp = Wbf + 3 * CD * CD;

  prep_w<<<dim3(CD * CD / 256, 4), 256, 0, stream>>>(q_w, k_w, v_w, proj_w, Wbf);

  dim3 gB(TD / 32, BB);
  branch_kernel<<<gB, 256, 0, stream>>>(queries, wq, q_b, q_g, q_bb, nullptr, qT, 0);
  branch_kernel<<<gB, 256, 0, stream>>>(keys,    wk, k_b, k_g, k_bb, mask,    kexp, 1);
  branch_kernel<<<gB, 256, 0, stream>>>(values,  wv, v_b, v_g, v_bb, nullptr, vfeat, 2);
  ctx_kernel<<<dim3(NH, BB), 128, 0, stream>>>(kexp, vfeat, ctxw);
  mmat_kernel<<<dim3(4, NH, BB), 256, 0, stream>>>(wp, ctxw, Mm);
  proj_kernel<<<gB, 256, 0, stream>>>(qT, Mm, proj_b, queries, norm_g, norm_b,
                                      (float*)d_out);
}
